// Quanvolution__gen110_65481071405836
// MI455X (gfx1250) — compile-verified, matrix operand order checked by reading, not run on hardware
//
#include <hip/hip_runtime.h>
#include <hip/hip_bf16.h>

typedef _Float16 v4h  __attribute__((ext_vector_type(4)));
typedef _Float16 v8h  __attribute__((ext_vector_type(8)));
typedef _Float16 v16h __attribute__((ext_vector_type(16)));
typedef float    v8f  __attribute__((ext_vector_type(8)));

#define NPIX   784
#define PITCH  808            // LDS row pitch in halves (16B aligned, 784 data + pad)
#define WAVES  4
#define TPB    (WAVES * 32)
#define RPW    16             // rows (samples) per wave
#define SPW    (WAVES * RPW)  // samples per workgroup = 64
#define NCLS   10

__global__ __launch_bounds__(TPB) void quanv_wmma_kernel(
    const float* __restrict__ x,      // (B,1,28,28)
    const float* __restrict__ theta,  // (4,)
    const float* __restrict__ Wc,     // (10,784)
    const float* __restrict__ bc,     // (10,)
    float* __restrict__ out)          // (B,10) log-softmax
{
  extern __shared__ char smem[];
  _Float16* FEAT = (_Float16*)smem;                    // WAVES*16*PITCH halves
  _Float16* WCS  = FEAT + WAVES * RPW * PITCH;         // 16*PITCH halves
  float*    CT   = (float*)(WCS + 16 * PITCH);         // WAVES*16*16 floats

  const int tid    = threadIdx.x;
  const int lane   = tid & 31;
  const int wave   = tid >> 5;
  const int wgBase = blockIdx.x * SPW;

  // ---- stage Wc -> LDS as f16, zero-padded to 16 rows x PITCH cols ----
  for (int idx = tid; idx < 16 * PITCH; idx += TPB) {
    int n = idx / PITCH, k = idx - n * PITCH;
    float v = (n < NCLS && k < NPIX) ? Wc[n * NPIX + k] : 0.0f;
    WCS[idx] = (_Float16)v;
  }
  // ---- zero the K-padding of the feature tiles (k in [784, PITCH)) ----
  for (int idx = tid; idx < WAVES * RPW * (PITCH - NPIX); idx += TPB) {
    int r = idx / (PITCH - NPIX);
    int k = NPIX + (idx - r * (PITCH - NPIX));
    FEAT[r * PITCH + k] = (_Float16)0.0f;
  }

  const float th0 = theta[0], th1 = theta[1], th2 = theta[2], th3 = theta[3];
  _Float16* feat    = FEAT + wave * RPW * PITCH;
  const int waveBase = wgBase + wave * RPW;

  // ---- phase 1: quantum features (cos + cumprod) -> LDS f16 ----
  // 16 rows * 196 patches = 3136 patch-slots per wave = 98 iters * 32 lanes
  for (int it = 0; it < 98; ++it) {
    int t   = it * 32 + lane;
    int row = t / 196;
    int p   = t - row * 196;
    int pi  = p / 14, pj = p - pi * 14;
    const float* px = x + (size_t)(waveBase + row) * NPIX + (2 * pi) * 28 + 2 * pj;
    __builtin_prefetch(px + 56, 0, 3);     // one patch-row ahead -> global_prefetch_b8
    float2 top = *(const float2*)px;         // pixels (2i,2j),(2i,2j+1)
    float2 bot = *(const float2*)(px + 28);  // pixels (2i+1,2j),(2i+1,2j+1)
    float c0 = __cosf(top.x + th0);
    float c1 = c0 * __cosf(top.y + th1);
    float c2 = c1 * __cosf(bot.x + th2);
    float c3 = c2 * __cosf(bot.y + th3);
    v4h hv = { (_Float16)c0, (_Float16)c1, (_Float16)c2, (_Float16)c3 };
    *(v4h*)(feat + row * PITCH + 4 * p) = hv;  // 8B-aligned ds_store_b64
  }
  __syncthreads();

  // ---- phase 2: WMMA GEMM, 25 chunks of K=32 (800 = 784 + pad) ----
  // f16 16x32 A layout: lanes 0-15 -> M=lane, K in {0..7, 16..23};
  //                     lanes 16-31 -> M=lane-16, K in {8..15, 24..31}.
  const int sub = lane >> 4;
  const int r16 = lane & 15;
  const _Float16* arow = feat + r16 * PITCH;
  const _Float16* brow = WCS  + r16 * PITCH;  // B mirrors A pattern, row = column N
  v8f acc = {};
  #pragma unroll 5
  for (int c = 0; c < 25; ++c) {
    int k0 = c * 32;
    v8h a_lo = *(const v8h*)(arow + k0 + sub * 8);        // ds_load_b128
    v8h a_hi = *(const v8h*)(arow + k0 + 16 + sub * 8);
    v8h b_lo = *(const v8h*)(brow + k0 + sub * 8);
    v8h b_hi = *(const v8h*)(brow + k0 + 16 + sub * 8);
    v16h A  = __builtin_shufflevector(a_lo, a_hi, 0,1,2,3,4,5,6,7,8,9,10,11,12,13,14,15);
    v16h Bm = __builtin_shufflevector(b_lo, b_hi, 0,1,2,3,4,5,6,7,8,9,10,11,12,13,14,15);
    acc = __builtin_amdgcn_wmma_f32_16x16x32_f16(
        /*neg_a=*/false, A, /*neg_b=*/false, Bm,
        /*c_mod=*/(short)0, acc, /*reuse_a=*/false, /*reuse_b=*/false);
  }

  // ---- phase 3: stage C tile, per-row log_softmax over 10 classes ----
  // C layout: VGPR r, lanes 0-15 -> (M=r, N=lane); lanes 16-31 -> (M=r+8, N=lane-16)
  float* ct = CT + wave * 256;
  #pragma unroll
  for (int r = 0; r < 8; ++r)
    ct[(r + sub * 8) * 16 + r16] = acc[r];
  __syncthreads();

  if (lane < 16) {
    int row    = lane;
    int sample = waveBase + row;
    float l[NCLS];
    float m = -1e30f;
    #pragma unroll
    for (int n = 0; n < NCLS; ++n) {
      l[n] = ct[row * 16 + n] + bc[n];
      m = fmaxf(m, l[n]);
    }
    float s = 0.0f;
    #pragma unroll
    for (int n = 0; n < NCLS; ++n) s += __expf(l[n] - m);
    float ls = __logf(s);
    #pragma unroll
    for (int n = 0; n < NCLS; ++n)
      out[(size_t)sample * NCLS + n] = l[n] - m - ls;
  }
}

extern "C" void kernel_launch(void* const* d_in, const int* in_sizes, int n_in,
                              void* d_out, int out_size, void* d_ws, size_t ws_size,
                              hipStream_t stream) {
  (void)in_sizes; (void)n_in; (void)out_size; (void)d_ws; (void)ws_size;
  const float* x     = (const float*)d_in[0];
  const float* theta = (const float*)d_in[1];
  const float* Wc    = (const float*)d_in[2];
  const float* bc    = (const float*)d_in[3];
  float* out = (float*)d_out;

  const size_t smem = (size_t)(WAVES * RPW * PITCH + 16 * PITCH) * sizeof(_Float16)
                    + (size_t)(WAVES * 256) * sizeof(float);   // ~130 KB, fits 320 KB WGP LDS
  (void)hipFuncSetAttribute((const void*)quanv_wmma_kernel,
                            hipFuncAttributeMaxDynamicSharedMemorySize, (int)smem);

  const int B = 16384;
  const int grid = B / SPW;  // 256 workgroups of 128 threads (4 waves)
  quanv_wmma_kernel<<<grid, TPB, smem, stream>>>(x, theta, Wc, bc, out);
}
